// MultiHeadAttentionBlock_1949915152886
// MI455X (gfx1250) — matrix operand order repaired and checked by reading, not compiled, not run on hardware
//
#include <hip/hip_runtime.h>
#include <hip/hip_bf16.h>
#include <cstdint>
#include <cstddef>

// Problem dims (compile-time constants from the reference)
#define B_  16
#define F_  1024
#define H_  16
#define D_  64
#define L_  8192
#define HD_ (H_ * D_)   // 1024

typedef __attribute__((ext_vector_type(16))) __bf16 v16bf;
typedef __attribute__((ext_vector_type(8)))  float  v8f;
typedef __attribute__((ext_vector_type(4)))  float  v4f;
typedef __attribute__((ext_vector_type(2)))  float  v2f;

// ---------------------------------------------------------------------------
// CDNA5 WMMA wrappers
// ---------------------------------------------------------------------------
__device__ __forceinline__ v8f wmma_bf16(v16bf a, v16bf b, v8f c) {
  return __builtin_amdgcn_wmma_f32_16x16x32_bf16(
      /*neg_a=*/false, a, /*neg_b=*/false, b,
      /*c_mod=*/(short)0, c, /*reuse_a=*/false, /*reuse_b=*/false);
}
// D(16x16 f32) = A(16x4 f32) * B(4x16 f32) + C — fp32-exact projections
__device__ __forceinline__ v8f wmma_f32(v2f a, v2f b, v8f c) {
  return __builtin_amdgcn_wmma_f32_16x16x4_f32(
      /*neg_a=*/false, a, /*neg_b=*/false, b,
      /*c_mod=*/(short)0, c, /*reuse_a=*/false, /*reuse_b=*/false);
}

// wave32 cross-lane helpers via ds_bpermute (index in bytes)
__device__ __forceinline__ float lane_bcast(float v, int src_lane) {
  return __int_as_float(
      __builtin_amdgcn_ds_bpermute(src_lane << 2, __float_as_int(v)));
}
__device__ __forceinline__ float wave_max32(float v, int lane) {
#pragma unroll
  for (int off = 16; off > 0; off >>= 1)
    v = fmaxf(v, __int_as_float(__builtin_amdgcn_ds_bpermute(
                     ((lane ^ off) & 31) << 2, __float_as_int(v))));
  return v;
}
__device__ __forceinline__ float wave_sum32(float v, int lane) {
#pragma unroll
  for (int off = 16; off > 0; off >>= 1)
    v += __int_as_float(__builtin_amdgcn_ds_bpermute(
        ((lane ^ off) & 31) << 2, __float_as_int(v)));
  return v;
}

__device__ __forceinline__ size_t kv_off(int b, int l, int h) {
  // [B, L, H, D] row-major, fp32
  return (((size_t)b * L_ + (size_t)l) * H_ + (size_t)h) * D_;
}

// ---------------------------------------------------------------------------
// fp32 GEMM: out[16,N] = A[16,K] @ W[K,N] + bias, one wave per 16-col tile.
// Uses V_WMMA_F32_16X16X4_F32 (fp32-exact, matches the fp32 reference math).
//
// A-matrix layout (32-bit, 16x4): lanes 0-15 hold M=0..15 with K=0 (V0),
//   K=1 (V1); lanes 16-31 hold K=2 (V0), K=3 (V1)  ->  a[e] = A[m][2*grp+e].
// B-matrix layout (4x16): lane's column N = lane&15; lanes 0-15 rows K=0,1;
//   lanes 16-31 rows K=2,3  ->  b[e] = W[k0+2*grp+e][n].
// C/D layout: VGPR r, lanes 0-15 -> M=r, lanes 16-31 -> M=r+8; N = lane&15.
// ---------------------------------------------------------------------------
__global__ __launch_bounds__(32) void gemm16_f32_kernel(
    const float* __restrict__ A, const float* __restrict__ W,
    const float* __restrict__ bias, float* __restrict__ out, int K, int N) {
  const int lane = threadIdx.x & 31;
  const int n16  = lane & 15;
  const int grp  = lane >> 4;
  const int n0   = blockIdx.x * 16;

  const float* arow = A + (size_t)n16 * K + grp * 2;          // this lane's A row
  const float* wcol = W + (size_t)(grp * 2) * N + n0 + n16;   // this lane's W col

  v8f c = {};
#pragma unroll 4
  for (int k0 = 0; k0 < K; k0 += 4) {
    v2f a;
    a[0] = arow[k0];
    a[1] = arow[k0 + 1];
    v2f bm;
    bm[0] = wcol[(size_t)k0 * N];
    bm[1] = wcol[(size_t)(k0 + 1) * N];
    c = wmma_f32(a, bm, c);
  }

  const float bv = bias[n0 + n16];
#pragma unroll
  for (int r = 0; r < 8; ++r)
    out[(size_t)(r + grp * 8) * N + n0 + n16] = c[r] + bv;
}

// ---------------------------------------------------------------------------
// Fused: KV ring-buffer update + full cache copy to d_out + decode attention.
// One workgroup (4 waves) per (b, h). Each wave owns L/4 contiguous positions,
// streams K/V rows from HBM exactly once with NT hints (read-once/write-once:
// keep the 192MB L2 for reused operands): the fp32 rows are copied verbatim to
// the output caches (substituting new k/v at the ring slot) while a bf16 copy
// feeds the QK^T WMMA and the online-softmax P*V accumulation.
// ---------------------------------------------------------------------------
__global__ __launch_bounds__(128) void attn_decode_kernel(
    const float* __restrict__ kin, const float* __restrict__ vin,
    const int* __restrict__ kv_idx,
    const float* __restrict__ qws, const float* __restrict__ kns,
    const float* __restrict__ vns,
    float* __restrict__ kout, float* __restrict__ vout,
    float* __restrict__ attn) {
  __shared__ float red_m[4], red_s[4], red_acc[4][64];

  const int bh   = blockIdx.x;          // 0..255
  const int b    = bh >> 4;
  const int h    = bh & 15;
  const int tid  = threadIdx.x;
  const int wave = tid >> 5;
  const int lane = tid & 31;
  const int n16  = lane & 15;
  const int grp  = lane >> 4;

  const int raw  = kv_idx[b];
  const int slot = raw % L_;                 // ring-buffer write slot
  const int seq  = min(raw + 1, L_);         // valid length after update

  const float* qp  = qws + (size_t)b * HD_ + h * D_;
  const float* knp = kns + (size_t)b * HD_ + h * D_;
  const float* vnp = vns + (size_t)b * HD_ + h * D_;

  // A operand: q (bf16) replicated across all 16 M rows; two K-chunks (D=64).
  v16bf a0, a1;
#pragma unroll
  for (int e = 0; e < 8; ++e) {
    a0[e]     = (__bf16)qp[grp * 16 + e];
    a0[e + 8] = (__bf16)qp[grp * 16 + 8 + e];
    a1[e]     = (__bf16)qp[32 + grp * 16 + e];
    a1[e + 8] = (__bf16)qp[32 + grp * 16 + 8 + e];
  }

  float m_run = -3.0e38f, s_run = 0.0f;
  float acc0 = 0.0f, acc1 = 0.0f;            // this lane owns d = 2*lane, 2*lane+1
  const int lbase  = wave * (L_ / 4);
  const int ntiles = L_ / (4 * 16);          // 128 tiles of 16 positions

  for (int t = 0; t < ntiles; ++t) {
    const int l0   = lbase + t * 16;
    const int lrow = l0 + n16;               // K row this lane loads (B operand col)

    if (t + 1 < ntiles) {                    // global_prefetch next K/V tile
      __builtin_prefetch(kin + kv_off(b, lrow + 16, h), 0, 1);
      __builtin_prefetch(vin + kv_off(b, lrow + 16, h), 0, 1);
    }

    // ---- K: stream fp32 row once (substitute new k at ring slot):
    //      NT copy to out cache + bf16 convert for the WMMA B operand.
    const size_t ko   = kv_off(b, lrow, h);
    const float* krow = (lrow == slot) ? knp : (kin + ko);
    float*       kdst = kout + ko;
    v16bf b0, b1;
#pragma unroll
    for (int i = 0; i < 4; ++i) {
      v4f t0 = __builtin_nontemporal_load(((const v4f*)(krow + grp * 16)) + i);
      __builtin_nontemporal_store(t0, ((v4f*)(kdst + grp * 16)) + i);
      b0[4 * i + 0] = (__bf16)t0[0]; b0[4 * i + 1] = (__bf16)t0[1];
      b0[4 * i + 2] = (__bf16)t0[2]; b0[4 * i + 3] = (__bf16)t0[3];
      v4f t1 = __builtin_nontemporal_load(((const v4f*)(krow + 32 + grp * 16)) + i);
      __builtin_nontemporal_store(t1, ((v4f*)(kdst + 32 + grp * 16)) + i);
      b1[4 * i + 0] = (__bf16)t1[0]; b1[4 * i + 1] = (__bf16)t1[1];
      b1[4 * i + 2] = (__bf16)t1[2]; b1[4 * i + 3] = (__bf16)t1[3];
    }

    // ---- logits for 16 positions via 2 WMMAs (wave-uniform branch: EXEC full)
    const bool compute = (l0 < seq);
    float p = 0.0f;
    if (compute) {
      v8f c = {};
      c = wmma_bf16(a0, b0, c);
      c = wmma_bf16(a1, b1, c);
      float logit = c[0] * 0.125f;           // 1/sqrt(D); lane's col = l0+n16
      const bool validl = (l0 + n16) < seq;
      logit = validl ? logit : -3.0e38f;

      const float tmax  = wave_max32(logit, lane);
      const float m_new = fmaxf(m_run, tmax);
      const float alpha = __expf(m_run - m_new);
      const float p_raw = validl ? __expf(logit - m_new) : 0.0f;
      p = (float)(__bf16)p_raw;              // probs are bf16 for P*V (ref)
      const float psum = wave_sum32(p_raw, lane);  // fp32 denominator (ref)
      s_run = s_run * alpha + psum;
      acc0 *= alpha;
      acc1 *= alpha;
      m_run = m_new;
    }

    // ---- V: stream rows once — NT copy to out cache + bf16 P*V accumulate
#pragma unroll 4
    for (int i = 0; i < 16; ++i) {
      const int    lv = l0 + i;
      const size_t vo = kv_off(b, lv, h);
      const float* vrow = (lv == slot) ? vnp : (vin + vo);
      v2f vv = __builtin_nontemporal_load(((const v2f*)vrow) + lane);
      __builtin_nontemporal_store(vv, ((v2f*)(vout + vo)) + lane);
      if (compute) {
        const float pi = lane_bcast(p, i);
        acc0 = fmaf(pi, (float)(__bf16)vv[0], acc0);
        acc1 = fmaf(pi, (float)(__bf16)vv[1], acc1);
      }
    }
  }

  // ---- combine the 4 wave-partials (m, s, acc[64]) through LDS
  if (lane == 0) { red_m[wave] = m_run; red_s[wave] = s_run; }
  red_acc[wave][lane * 2]     = acc0;
  red_acc[wave][lane * 2 + 1] = acc1;
  __syncthreads();

  if (tid < 64) {
    const float m = fmaxf(fmaxf(red_m[0], red_m[1]), fmaxf(red_m[2], red_m[3]));
    float s = 0.0f, a = 0.0f;
#pragma unroll
    for (int w = 0; w < 4; ++w) {
      const float sc = __expf(red_m[w] - m);
      s = fmaf(red_s[w], sc, s);
      a = fmaf(red_acc[w][tid], sc, a);
    }
    attn[(size_t)b * HD_ + h * D_ + tid] = a / s;
  }
}

__global__ void newidx_kernel(const int* __restrict__ kv_idx,
                              int* __restrict__ nout) {
  const int i = threadIdx.x;
  if (i < B_) nout[i] = kv_idx[i] + 1;
}

// ---------------------------------------------------------------------------
// Host-side launcher
// ---------------------------------------------------------------------------
extern "C" void kernel_launch(void* const* d_in, const int* in_sizes, int n_in,
                              void* d_out, int out_size, void* d_ws,
                              size_t ws_size, hipStream_t stream) {
  (void)in_sizes; (void)n_in; (void)out_size; (void)ws_size;

  const float* x      = (const float*)d_in[0];
  const float* kv_key = (const float*)d_in[1];
  const float* kv_val = (const float*)d_in[2];
  const int*   kv_idx = (const int*)d_in[3];
  const float* Wq = (const float*)d_in[4];
  const float* bq = (const float*)d_in[5];
  const float* Wk = (const float*)d_in[6];
  const float* bk = (const float*)d_in[7];
  const float* Wv = (const float*)d_in[8];
  const float* bv = (const float*)d_in[9];
  const float* Wo = (const float*)d_in[10];
  const float* bo = (const float*)d_in[11];

  // Outputs concatenated flat: y[B,F], kv_key[B,L,H,D], kv_value[B,L,H,D], new_idx[B]
  float* out   = (float*)d_out;
  float* y_out = out;
  float* k_out = out + (size_t)B_ * F_;
  float* v_out = k_out + (size_t)B_ * L_ * HD_;
  int*   i_out = (int*)(v_out + (size_t)B_ * L_ * HD_);

  // Scratch: q, k_new, v_new, attn — each [B, H*D] fp32 (64 KB each)
  float* ws   = (float*)d_ws;
  float* q_ws = ws;
  float* k_ws = ws + 1 * (size_t)B_ * HD_;
  float* v_ws = ws + 2 * (size_t)B_ * HD_;
  float* a_ws = ws + 3 * (size_t)B_ * HD_;

  // QKV projections (fp32-exact WMMA): [16,1024] @ [1024,1024] each
  gemm16_f32_kernel<<<dim3(HD_ / 16), 32, 0, stream>>>(x, Wq, bq, q_ws, F_, HD_);
  gemm16_f32_kernel<<<dim3(HD_ / 16), 32, 0, stream>>>(x, Wk, bk, k_ws, F_, HD_);
  gemm16_f32_kernel<<<dim3(HD_ / 16), 32, 0, stream>>>(x, Wv, bv, v_ws, F_, HD_);

  // Fused KV-update + cache copy + decode attention: one WG per (b,h)
  attn_decode_kernel<<<dim3(B_ * H_), 128, 0, stream>>>(
      kv_key, kv_val, kv_idx, q_ws, k_ws, v_ws, k_out, v_out, a_ws);

  // Output projection (fp32-exact WMMA): [16,1024] @ [1024,1024]
  gemm16_f32_kernel<<<dim3(F_ / 16), 32, 0, stream>>>(a_ws, Wo, bo, y_out, HD_, F_);

  // new_idx = kv_idx + 1
  newidx_kernel<<<1, 32, 0, stream>>>(kv_idx, i_out);
}
